// MambaModel_68556267979149
// MI455X (gfx1250) — compile-verified, matrix operand order checked by reading, not run on hardware
//
#include <hip/hip_runtime.h>
#include <hip/hip_bf16.h>

#define BATCH 8192
#define SEQN  16
#define IND   512
#define HD    256
#define DSD   64
#define NL    6
#define NCLS  40

#define YS (HD + 8)       // padded LDS stride for LN output (f16)
#define HS (DSD + 8)      // padded LDS stride for scan output (f16)
#define FS (2 * HD + 8)   // padded LDS stride for FFN hidden / staged input (f16)

// f16 weight cache layout inside d_ws (element offsets, _Float16)
#define OFF_INW 0u
#define OFF_A   131072u
#define OFF_B   229376u
#define OFF_C   327680u
#define OFF_D   425984u
#define OFF_G   819200u
#define OFF_W1  1212416u
#define OFF_W2  1998848u

typedef __attribute__((ext_vector_type(16))) _Float16 v16h;
typedef __attribute__((ext_vector_type(8)))  float    v8f;

// ---- one-shot fp32 -> f16 weight conversion into d_ws ---------------------
__global__ __launch_bounds__(256) void cvt_f32_to_f16(const float* __restrict__ s,
                                                      _Float16* __restrict__ d, int n) {
  const int i = (blockIdx.x * 256 + threadIdx.x) * 4;
  if (i < n) {
    const float4 v = *(const float4*)(s + i);
    d[i + 0] = (_Float16)v.x; d[i + 1] = (_Float16)v.y;
    d[i + 2] = (_Float16)v.z; d[i + 3] = (_Float16)v.w;
  }
}

// ---- WMMA fragment loaders (CDNA5 16-bit A/B layout, wave32) --------------
// A (16x32 slice): lane<16 -> M=lane, K = k0+{0..7,16..23};
// lane>=16 -> M=lane-16, K = k0+{8..15,24..31}. Two contiguous 8-half runs.
__device__ __forceinline__ v16h frag_a_lds(const _Float16* base, int stride,
                                           int k0, int lane) {
  const int m  = lane & 15;
  const int kb = k0 + ((lane & 16) ? 8 : 0);
  const _Float16* p = base + m * stride + kb;
  v16h f;
#pragma unroll
  for (int i = 0; i < 16; ++i) f[i] = p[((i & 8) << 1) + (i & 7)];
  return f;
}

// B fragment from f16 weights W row-major [N][K]: two 16B runs -> 2x global_load_b128
__device__ __forceinline__ v16h frag_b_h16(const _Float16* __restrict__ W, int ldk,
                                           int n0, int k0, int lane) {
  const int n  = n0 + (lane & 15);
  const int kb = k0 + ((lane & 16) ? 8 : 0);
  const _Float16* p = W + (size_t)n * ldk + kb;
  v16h f;
#pragma unroll
  for (int i = 0; i < 8; ++i)  f[i] = p[i];
#pragma unroll
  for (int i = 8; i < 16; ++i) f[i] = p[i + 8];
  return f;
}

// ---- fp32 LayerNorm over H=256 for 16 tokens: 16 lanes per token ----------
__device__ __forceinline__ void ln16(const float* __restrict__ xr,
                                     _Float16* __restrict__ y, int ys,
                                     const float* __restrict__ g,
                                     const float* __restrict__ b, int tid) {
  const int tk = tid >> 4, l = tid & 15;
  float s1 = 0.f, s2 = 0.f;
  for (int j = l; j < HD; j += 16) { const float v = xr[tk * HD + j]; s1 += v; s2 += v * v; }
#pragma unroll
  for (int m = 8; m; m >>= 1) { s1 += __shfl_xor(s1, m, 16); s2 += __shfl_xor(s2, m, 16); }
  const float mu = s1 * (1.f / HD);
  const float rs = rsqrtf(s2 * (1.f / HD) - mu * mu + 1e-5f);
  for (int j = l; j < HD; j += 16)
    y[tk * ys + j] = (_Float16)((xr[tk * HD + j] - mu) * rs * g[j] + b[j]);
}

// ---- whole network, one workgroup per batch row ---------------------------
__global__ __launch_bounds__(256) void mamba_fused(
    const float* __restrict__ data,
    const _Float16* __restrict__ wsh,          // f16 weight cache (d_ws)
    const float* __restrict__ in_b,
    const float* __restrict__ norm_g, const float* __restrict__ norm_b,
    const float* __restrict__ D_b,    const float* __restrict__ gate_b,
    const float* __restrict__ ffn_b1, const float* __restrict__ ffn_b2,
    const float* __restrict__ fln_g,  const float* __restrict__ fln_b,
    const float* __restrict__ out_w,  const float* __restrict__ out_b,
    const float* __restrict__ cls_w,  const float* __restrict__ cls_b,
    const float* __restrict__ attn_w, const float* __restrict__ attn_b,
    float* __restrict__ out) {
  __shared__ __align__(16) float    xres[SEQN][HD];   // fp32 residual (16 KB)
  __shared__ __align__(16) _Float16 yln[SEQN][YS];    // LN output
  __shared__ __align__(16) _Float16 sa[SEQN][DSD];    // tanh(A proj)
  __shared__ __align__(16) _Float16 sb[SEQN][DSD];    // B proj
  __shared__ __align__(16) _Float16 sg[SEQN][HD];     // sigmoid gate
  __shared__ __align__(16) _Float16 sdt[SEQN][HD];    // D proj
  __shared__ __align__(16) _Float16 hsc[SEQN][HS];    // scan states
  __shared__ __align__(16) _Float16 ffh[SEQN][FS];    // FFN hidden / staged input

  const int tid  = threadIdx.x;
  const int lane = tid & 31;
  const int wave = tid >> 5;
  const size_t row0 = (size_t)blockIdx.x * SEQN * IND;

  // stage this row's raw input as f16 into ffh (A operand of the input GEMM)
  for (int i = tid; i < SEQN * (IND / 4); i += 256) {
    const int r = i / (IND / 4), c = (i % (IND / 4)) * 4;
    const float4 v = *(const float4*)(data + row0 + (size_t)r * IND + c);
    ffh[r][c + 0] = (_Float16)v.x; ffh[r][c + 1] = (_Float16)v.y;
    ffh[r][c + 2] = (_Float16)v.z; ffh[r][c + 3] = (_Float16)v.w;
  }
  __syncthreads();

  // input projection: x = data @ in_w.T + in_b  (16x512 @ 512x256), 2 tiles/wave
  {
    const _Float16* Winw = wsh + OFF_INW;
    v8f acc0 = {}, acc1 = {};
    const int nt0 = wave, nt1 = wave + 8;
    for (int k0 = 0; k0 < IND; k0 += 32) {
      const v16h a  = frag_a_lds(&ffh[0][0], FS, k0, lane);
      const v16h b0 = frag_b_h16(Winw, IND, nt0 * 16, k0, lane);
      acc0 = __builtin_amdgcn_wmma_f32_16x16x32_f16(false, a, false, b0, (short)0, acc0, false, false);
      const v16h b1 = frag_b_h16(Winw, IND, nt1 * 16, k0, lane);
      acc1 = __builtin_amdgcn_wmma_f32_16x16x32_f16(false, a, false, b1, (short)0, acc1, false, false);
    }
    const int mb = (lane & 16) ? 8 : 0;
    const int n0 = nt0 * 16 + (lane & 15), n1 = nt1 * 16 + (lane & 15);
    const float bi0 = in_b[n0], bi1 = in_b[n1];
#pragma unroll
    for (int r = 0; r < 8; ++r) {
      xres[mb + r][n0] = acc0[r] + bi0;
      xres[mb + r][n1] = acc1[r] + bi1;
    }
  }
  __syncthreads();

  for (int l = 0; l < NL; ++l) {
    const float* ng = norm_g + l * HD;  const float* nb = norm_b + l * HD;
    const float* Db = D_b + l * HD;     const float* gb = gate_b + l * HD;
    const float* b1 = ffn_b1 + l * 2 * HD;
    const float* b2 = ffn_b2 + l * HD;
    const _Float16* Aw = wsh + OFF_A  + (size_t)l * DSD * HD;
    const _Float16* Bw = wsh + OFF_B  + (size_t)l * DSD * HD;
    const _Float16* Cw = wsh + OFF_C  + (size_t)l * HD * DSD;
    const _Float16* Dw = wsh + OFF_D  + (size_t)l * HD * HD;
    const _Float16* gw = wsh + OFF_G  + (size_t)l * HD * HD;
    const _Float16* w1 = wsh + OFF_W1 + (size_t)l * 2 * HD * HD;
    const _Float16* w2 = wsh + OFF_W2 + (size_t)l * HD * 2 * HD;

    ln16(&xres[0][0], &yln[0][0], YS, ng, nb, tid);
    __syncthreads();

    // A,B (256->64) + gate,D (256->256): 40 N-tiles, K=256; 5 tiles/wave,
    // k-outer so the shared A fragment is loaded once per k-step.
    {
      const _Float16* Wt[5]; int ntv[5], kindv[5];
#pragma unroll
      for (int j = 0; j < 5; ++j) {
        const int t = wave + 8 * j;
        if (t < 4)       { Wt[j] = Aw; ntv[j] = t;      kindv[j] = 0; }
        else if (t < 8)  { Wt[j] = Bw; ntv[j] = t - 4;  kindv[j] = 1; }
        else if (t < 24) { Wt[j] = gw; ntv[j] = t - 8;  kindv[j] = 2; }
        else             { Wt[j] = Dw; ntv[j] = t - 24; kindv[j] = 3; }
      }
      v8f acc[5];
#pragma unroll
      for (int j = 0; j < 5; ++j) acc[j] = (v8f){};
      for (int k0 = 0; k0 < HD; k0 += 32) {
        const v16h a = frag_a_lds(&yln[0][0], YS, k0, lane);
#pragma unroll
        for (int j = 0; j < 5; ++j) {
          const v16h b = frag_b_h16(Wt[j], HD, ntv[j] * 16, k0, lane);
          acc[j] = __builtin_amdgcn_wmma_f32_16x16x32_f16(false, a, false, b, (short)0, acc[j], false, false);
        }
      }
      const int mb = (lane & 16) ? 8 : 0;
#pragma unroll
      for (int j = 0; j < 5; ++j) {
        const int n = ntv[j] * 16 + (lane & 15);
        const int kind = kindv[j];
#pragma unroll
        for (int r = 0; r < 8; ++r) {
          const int m = mb + r;
          const float v = acc[j][r];
          if (kind == 0)      sa[m][n]  = (_Float16)tanhf(v);
          else if (kind == 1) sb[m][n]  = (_Float16)v;
          else if (kind == 2) sg[m][n]  = (_Float16)(1.f / (1.f + expf(-(v + gb[n]))));
          else                sdt[m][n] = (_Float16)(v + Db[n]);
        }
      }
    }
    __syncthreads();

    // sequential scan h_s = a_s*h + b_s over the 16 tokens (fp32 recurrence)
    if (tid < DSD) {
      float h = 0.f;
#pragma unroll
      for (int s = 0; s < SEQN; ++s) {
        h = (float)sa[s][tid] * h + (float)sb[s][tid];
        hsc[s][tid] = (_Float16)h;
      }
    }
    __syncthreads();

    // ys = hs @ Cw.T (K=64); gated residual x += ys*g + dt*(1-g); 2 tiles/wave
    {
      v8f acc0 = {}, acc1 = {};
      const int nt0 = wave, nt1 = wave + 8;
      for (int k0 = 0; k0 < DSD; k0 += 32) {
        const v16h a  = frag_a_lds(&hsc[0][0], HS, k0, lane);
        const v16h b0 = frag_b_h16(Cw, DSD, nt0 * 16, k0, lane);
        acc0 = __builtin_amdgcn_wmma_f32_16x16x32_f16(false, a, false, b0, (short)0, acc0, false, false);
        const v16h b1 = frag_b_h16(Cw, DSD, nt1 * 16, k0, lane);
        acc1 = __builtin_amdgcn_wmma_f32_16x16x32_f16(false, a, false, b1, (short)0, acc1, false, false);
      }
      const int mb = (lane & 16) ? 8 : 0;
      const int n0 = nt0 * 16 + (lane & 15), n1 = nt1 * 16 + (lane & 15);
#pragma unroll
      for (int r = 0; r < 8; ++r) {
        const int m = mb + r;
        float gf = (float)sg[m][n0];
        xres[m][n0] += acc0[r] * gf + (float)sdt[m][n0] * (1.f - gf);
        gf = (float)sg[m][n1];
        xres[m][n1] += acc1[r] * gf + (float)sdt[m][n1] * (1.f - gf);
      }
    }
    __syncthreads();

    // FFN: LN -> 256->512 GELU(exact) -> 512->256 residual
    ln16(&xres[0][0], &yln[0][0], YS, ng, nb, tid);
    __syncthreads();
    {  // FFN1: 32 tiles, 4 per wave, k-outer
      v8f acc[4];
#pragma unroll
      for (int j = 0; j < 4; ++j) acc[j] = (v8f){};
      for (int k0 = 0; k0 < HD; k0 += 32) {
        const v16h a = frag_a_lds(&yln[0][0], YS, k0, lane);
#pragma unroll
        for (int j = 0; j < 4; ++j) {
          const v16h b = frag_b_h16(w1, HD, (wave + 8 * j) * 16, k0, lane);
          acc[j] = __builtin_amdgcn_wmma_f32_16x16x32_f16(false, a, false, b, (short)0, acc[j], false, false);
        }
      }
      const int mb = (lane & 16) ? 8 : 0;
#pragma unroll
      for (int j = 0; j < 4; ++j) {
        const int n = (wave + 8 * j) * 16 + (lane & 15);
        const float bias = b1[n];
#pragma unroll
        for (int r = 0; r < 8; ++r) {
          const float v = acc[j][r] + bias;
          ffh[mb + r][n] = (_Float16)(0.5f * v * (1.f + erff(v * 0.70710678f)));
        }
      }
    }
    __syncthreads();
    {  // FFN2: 16 tiles, 2 per wave, K=512
      v8f acc0 = {}, acc1 = {};
      const int nt0 = wave, nt1 = wave + 8;
      for (int k0 = 0; k0 < 2 * HD; k0 += 32) {
        const v16h a  = frag_a_lds(&ffh[0][0], FS, k0, lane);
        const v16h b0 = frag_b_h16(w2, 2 * HD, nt0 * 16, k0, lane);
        acc0 = __builtin_amdgcn_wmma_f32_16x16x32_f16(false, a, false, b0, (short)0, acc0, false, false);
        const v16h b1 = frag_b_h16(w2, 2 * HD, nt1 * 16, k0, lane);
        acc1 = __builtin_amdgcn_wmma_f32_16x16x32_f16(false, a, false, b1, (short)0, acc1, false, false);
      }
      const int mb = (lane & 16) ? 8 : 0;
      const int n0 = nt0 * 16 + (lane & 15), n1 = nt1 * 16 + (lane & 15);
      const float bi0 = b2[n0], bi1 = b2[n1];
#pragma unroll
      for (int r = 0; r < 8; ++r) {
        xres[mb + r][n0] += acc0[r] + bi0;
        xres[mb + r][n1] += acc1[r] + bi1;
      }
    }
    __syncthreads();
  }

  // ---- final LN + attention pooling + heads + log_softmax -----------------
  ln16(&xres[0][0], &yln[0][0], YS, fln_g, fln_b, tid);
  __syncthreads();

  float* fbuf   = reinterpret_cast<float*>(&sg[0][0]); // sg dead; 2048 floats
  float* sc     = fbuf;                       // [16]
  float* pool   = fbuf + 16;                  // [256]
  float* hid    = fbuf + 16 + 256;            // [128]
  float* logits = fbuf + 16 + 256 + 128;      // [40]
  float* tmp    = fbuf + 16 + 256 + 128 + 64; // [2]

  {  // score_s = dot(node, w[0:H]) + dot(nbr_s, w[H:2H]) (+ bias later)
    const int tk = tid >> 4, l = tid & 15;
    float p = 0.f;
    const float* wv = (tk == 0) ? attn_w : (attn_w + HD);
    for (int j = l; j < HD; j += 16) p += (float)yln[tk][j] * wv[j];
#pragma unroll
    for (int m = 8; m; m >>= 1) p += __shfl_xor(p, m, 16);
    if (l == 0) sc[tk] = p;
  }
  __syncthreads();
  if (tid == 0) {  // softmax over 15 neighbors
    const float cn = sc[0] + attn_b[0];
    float mx = -3.4e38f;
    for (int s = 1; s < SEQN; ++s) mx = fmaxf(mx, cn + sc[s]);
    float den = 0.f;
    for (int s = 1; s < SEQN; ++s) { const float e = expf(cn + sc[s] - mx); sc[s] = e; den += e; }
    const float rden = 1.f / den;
    for (int s = 1; s < SEQN; ++s) sc[s] *= rden;
  }
  __syncthreads();
  if (tid < HD) {  // node + weighted neighbor sum
    float p = (float)yln[0][tid];
    for (int s = 1; s < SEQN; ++s) p += (float)yln[s][tid] * sc[s];
    pool[tid] = p;
  }
  __syncthreads();
  if (tid < HD / 2) {  // relu(out_w @ pool + out_b)
    float acc = out_b[tid];
    const float* wr = out_w + (size_t)tid * HD;
    for (int j = 0; j < HD; ++j) acc += pool[j] * wr[j];
    hid[tid] = fmaxf(acc, 0.f);
  }
  __syncthreads();
  if (tid < NCLS) {  // classifier
    float acc = cls_b[tid];
    const float* wr = cls_w + (size_t)tid * (HD / 2);
    for (int j = 0; j < HD / 2; ++j) acc += hid[j] * wr[j];
    logits[tid] = acc;
  }
  __syncthreads();
  if (tid == 0) {
    float mx = -3.4e38f;
    for (int c = 0; c < NCLS; ++c) mx = fmaxf(mx, logits[c]);
    float s = 0.f;
    for (int c = 0; c < NCLS; ++c) s += expf(logits[c] - mx);
    tmp[0] = mx; tmp[1] = logf(s);
  }
  __syncthreads();
  if (tid < NCLS)
    out[(size_t)blockIdx.x * NCLS + tid] = logits[tid] - tmp[0] - tmp[1];
}

extern "C" void kernel_launch(void* const* d_in, const int* in_sizes, int n_in,
                              void* d_out, int out_size, void* d_ws, size_t ws_size,
                              hipStream_t stream) {
  const float* data   = (const float*)d_in[0];
  const float* in_w   = (const float*)d_in[1];
  const float* in_b   = (const float*)d_in[2];
  const float* norm_g = (const float*)d_in[3];
  const float* norm_b = (const float*)d_in[4];
  const float* A_w    = (const float*)d_in[5];
  const float* B_w    = (const float*)d_in[6];
  const float* C_w    = (const float*)d_in[7];
  const float* D_w    = (const float*)d_in[8];
  const float* D_b    = (const float*)d_in[9];
  const float* gate_w = (const float*)d_in[10];
  const float* gate_b = (const float*)d_in[11];
  const float* ffn_w1 = (const float*)d_in[12];
  const float* ffn_b1 = (const float*)d_in[13];
  const float* ffn_w2 = (const float*)d_in[14];
  const float* ffn_b2 = (const float*)d_in[15];
  const float* fln_g  = (const float*)d_in[16];
  const float* fln_b  = (const float*)d_in[17];
  const float* out_w  = (const float*)d_in[18];
  const float* out_b  = (const float*)d_in[19];
  const float* cls_w  = (const float*)d_in[20];
  const float* cls_b  = (const float*)d_in[21];
  const float* attn_w = (const float*)d_in[22];
  const float* attn_b = (const float*)d_in[23];

  _Float16* wsh = (_Float16*)d_ws;

  // one-shot weight down-conversion into the f16 cache (5.6 MB in d_ws)
  struct { const float* s; unsigned off; int n; } cv[8] = {
    { in_w,   OFF_INW, HD * IND },
    { A_w,    OFF_A,   NL * DSD * HD },
    { B_w,    OFF_B,   NL * DSD * HD },
    { C_w,    OFF_C,   NL * HD * DSD },
    { D_w,    OFF_D,   NL * HD * HD },
    { gate_w, OFF_G,   NL * HD * HD },
    { ffn_w1, OFF_W1,  NL * 2 * HD * HD },
    { ffn_w2, OFF_W2,  NL * HD * 2 * HD },
  };
  for (int i = 0; i < 8; ++i) {
    const int blocks = (cv[i].n / 4 + 255) / 256;
    cvt_f32_to_f16<<<dim3(blocks), dim3(256), 0, stream>>>(cv[i].s, wsh + cv[i].off, cv[i].n);
  }

  mamba_fused<<<dim3(BATCH), dim3(256), 0, stream>>>(
      data, wsh, in_b, norm_g, norm_b, D_b, gate_b, ffn_b1, ffn_b2,
      fln_g, fln_b, out_w, out_b, cls_w, cls_b, attn_w, attn_b, (float*)d_out);
}
